// NoisyTopKRouter_29738353557800
// MI455X (gfx1250) — matrix operand order repaired and checked by reading, not compiled, not run on hardware
//
#include <hip/hip_runtime.h>
#include <hip/hip_bf16.h>
#include <math.h>

// Problem constants (from reference)
#define B_ 4
#define S_ 4096
#define D_ 2048
#define E_ 64
#define K_ 8
#define NTOK (B_ * S_)

typedef __attribute__((ext_vector_type(16))) __bf16 v16bf;
typedef __attribute__((ext_vector_type(16))) unsigned short us16;
typedef __attribute__((ext_vector_type(8))) float v8f;

__device__ __forceinline__ unsigned short f2bf(float f) {
    // round-to-nearest-even f32 -> bf16 (used only in prep kernel)
    unsigned int u = __float_as_uint(f);
    unsigned int r = u + 0x7FFFu + ((u >> 16) & 1u);
    return (unsigned short)(r >> 16);
}

__device__ __forceinline__ float softplus_f(float v) {
    // stable & cheap: max(v,0) + log(1 + exp(-|v|)); hardware exp/log
    float e = __expf(-fabsf(v));
    return fmaxf(v, 0.f) + __logf(1.f + e);
}

// ---------------------------------------------------------------------------
// Kernel 0: convert Wr/Wn to bf16 in workspace; write new_bias (deterministic:
// avg_load - total_selections < 0 always, computed faithfully).
// ---------------------------------------------------------------------------
__global__ void prep_kernel(const float* __restrict__ Wr,
                            const float* __restrict__ Wn,
                            const float* __restrict__ bias,
                            unsigned short* __restrict__ wrb,
                            unsigned short* __restrict__ wnb,
                            float* __restrict__ new_bias) {
    int i = blockIdx.x * blockDim.x + threadIdx.x;
    if (i < E_ * D_) {
        wrb[i] = f2bf(Wr[i]);
        wnb[i] = f2bf(Wn[i]);
    }
    if (i < E_) {
        const float total = (float)(B_ * S_ * K_);
        const float avg = total / (float)E_;
        float lv = avg - total;
        float sg = (lv > 0.f) ? 1.f : ((lv < 0.f) ? -1.f : 0.f);
        new_bias[i] = bias[i] + 0.001f * sg;
    }
}

// Convert 4x float4 (A-chunk, contiguous K order, same as B) into one v16bf frag
__device__ __forceinline__ v16bf cvtA(const float4 f[4]) {
    v16bf a;
#pragma unroll
    for (int i = 0; i < 4; ++i) {
        a[4 * i + 0] = (__bf16)f[i].x;
        a[4 * i + 1] = (__bf16)f[i].y;
        a[4 * i + 2] = (__bf16)f[i].z;
        a[4 * i + 3] = (__bf16)f[i].w;
    }
    return a;
}

// ---------------------------------------------------------------------------
// Kernel 1: fused dual-GEMM (WMMA bf16) + noisy top-k router.
// Block = 128 threads (4 waves). Each wave: 16 tokens x 64 experts.
// K-loop unrolled x2 with ping-pong A buffers (no register rotation).
// ---------------------------------------------------------------------------
__global__ __launch_bounds__(128) void router_kernel(
    const float* __restrict__ x,
    const float* __restrict__ bias,
    const float* __restrict__ noise_u,
    const unsigned short* __restrict__ wrb,
    const unsigned short* __restrict__ wnb,
    float* __restrict__ probs,
    float* __restrict__ topkf) {

    __shared__ __align__(16) float zbuf[4][16 * 68];  // padded rows: bank-conflict-free
    __shared__ float bias_sh[E_];

    const int tid  = threadIdx.x;
    const int wid  = tid >> 5;
    const int lane = tid & 31;
    const int hrow = lane & 15;  // M (A/C rows) or N (B/C cols)
    const int kg   = lane >> 4;  // lane half selector

    if (tid < E_) bias_sh[tid] = bias[tid];
    __syncthreads();

    const int block_tok = blockIdx.x * 64;
    const int wave_tok  = block_tok + wid * 16;
    const int my_tok    = wave_tok + hrow;  // A-matrix row owned by this lane

    v8f accr[4] = {v8f{}, v8f{}, v8f{}, v8f{}};
    v8f accn[4] = {v8f{}, v8f{}, v8f{}, v8f{}};

    const float* xrow = x + (size_t)my_tok * D_;

    // A-chunk loader: contiguous K order (matches B):
    // lanes 0-15 K={k0..k0+15}; lanes 16-31 K={k0+16..k0+31}
    auto loadA = [&](int k0, float4 f[4]) {
        const float4* p0 = reinterpret_cast<const float4*>(xrow + k0 + kg * 16);
        const float4* p1 = reinterpret_cast<const float4*>(xrow + k0 + kg * 16 + 8);
        f[0] = p0[0]; f[1] = p0[1]; f[2] = p1[0]; f[3] = p1[1];
    };
    // B-fragment loader (32x16 bf16): lane holds 16 consecutive K for col e
    auto loadB = [&](int k0, us16 br[4], us16 bn[4]) {
#pragma unroll
        for (int nt = 0; nt < 4; ++nt) {
            const int e = nt * 16 + hrow;
            const size_t woff = (size_t)e * D_ + (size_t)(k0 + kg * 16);
            br[nt] = *reinterpret_cast<const us16*>(wrb + woff);
            bn[nt] = *reinterpret_cast<const us16*>(wnb + woff);
        }
    };
    auto mm8 = [&](v16bf a, const us16 br[4], const us16 bn[4]) {
#pragma unroll
        for (int nt = 0; nt < 4; ++nt) {
            accr[nt] = __builtin_amdgcn_wmma_f32_16x16x32_bf16(
                false, a, false, __builtin_bit_cast(v16bf, br[nt]),
                (short)0, accr[nt], false, false);
        }
#pragma unroll
        for (int nt = 0; nt < 4; ++nt) {
            accn[nt] = __builtin_amdgcn_wmma_f32_16x16x32_bf16(
                false, a, false, __builtin_bit_cast(v16bf, bn[nt]),
                (short)0, accn[nt], false, false);
        }
    };

    // ---- Software pipeline, 2x unrolled, ping-pong A buffers
    float4 fA0[4], fA1[4];
    loadA(0, fA0);
    loadA(32, fA1);

    for (int k0 = 0; k0 < D_; k0 += 64) {
        // half 1: uses fA0; prefetch A for k0+64 into fA0
        us16 br0[4], bn0[4];
        loadB(k0, br0, bn0);
        v16bf a0 = cvtA(fA0);
        loadA((k0 + 64 < D_) ? (k0 + 64) : 0, fA0);  // clamp: harmless re-read on last iter
        mm8(a0, br0, bn0);

        // half 2: uses fA1; prefetch A for k0+96 into fA1
        us16 br1[4], bn1[4];
        loadB(k0 + 32, br1, bn1);
        v16bf a1 = cvtA(fA1);
        loadA((k0 + 96 < D_) ? (k0 + 96) : 0, fA1);
        mm8(a1, br1, bn1);
    }

    // ---- Epilogue: z = noise_u * softplus(noisy) + logits + bias -> LDS
    // C/D layout: VGPR r -> M = r + 8*(lane/16), N = lane%16
#pragma unroll
    for (int nt = 0; nt < 4; ++nt) {
        const int e = nt * 16 + hrow;
#pragma unroll
        for (int r = 0; r < 8; ++r) {
            int m = r + 8 * kg;
            int tok = wave_tok + m;
            float sp = softplus_f(accn[nt][r]);
            float z  = noise_u[(size_t)tok * E_ + e] * sp + accr[nt][r] + bias_sh[e];
            zbuf[wid][m * 68 + e] = z;
        }
    }
    __syncthreads();

    // ---- Top-8 + sparse softmax: one lane per token (lanes 0..15 of each wave)
    if (lane < 16) {
        float* row = &zbuf[wid][lane * 68];
        float topv[K_];
        int   topi[K_];
#pragma unroll
        for (int k = 0; k < K_; ++k) {
            float best = -__builtin_inff();
            int bi = 0;
#pragma unroll 8
            for (int j = 0; j < E_; ++j) {
                float v = row[j];
                if (v > best) { best = v; bi = j; }
            }
            topv[k] = best;
            topi[k] = bi;
            row[bi] = -__builtin_inff();
        }
        float mx = topv[0];
        float ex[K_];
        float s = 0.f;
#pragma unroll
        for (int k = 0; k < K_; ++k) { ex[k] = __expf(topv[k] - mx); s += ex[k]; }
        float inv = 1.f / s;

        // probs row: zero then scatter top-k
        float4* row4 = reinterpret_cast<float4*>(row);
#pragma unroll
        for (int j = 0; j < 16; ++j) row4[j] = make_float4(0.f, 0.f, 0.f, 0.f);
#pragma unroll
        for (int k = 0; k < K_; ++k) row[topi[k]] = ex[k] * inv;

        // top-k indices (as float), descending-value order == scan order
        int tok = wave_tok + lane;
        float4 i0 = make_float4((float)topi[0], (float)topi[1], (float)topi[2], (float)topi[3]);
        float4 i1 = make_float4((float)topi[4], (float)topi[5], (float)topi[6], (float)topi[7]);
        float4* tout = reinterpret_cast<float4*>(topkf + (size_t)tok * K_);
        tout[0] = i0;
        tout[1] = i1;
    }
    __syncthreads();

    // ---- Coalesced probs store: 64 tokens * 64 experts per block
    float* pout = probs + (size_t)block_tok * E_;
#pragma unroll
    for (int i = 0; i < 8; ++i) {
        int flat = tid * 4 + i * 512;
        int t = flat >> 6;
        int e = flat & 63;
        float4 v = *reinterpret_cast<const float4*>(&zbuf[t >> 4][(t & 15) * 68 + e]);
        *reinterpret_cast<float4*>(pout + flat) = v;
    }
}

// ---------------------------------------------------------------------------
extern "C" void kernel_launch(void* const* d_in, const int* in_sizes, int n_in,
                              void* d_out, int out_size, void* d_ws, size_t ws_size,
                              hipStream_t stream) {
    const float* x       = (const float*)d_in[0];
    const float* Wr      = (const float*)d_in[1];
    const float* Wn      = (const float*)d_in[2];
    const float* bias    = (const float*)d_in[3];
    const float* noise_u = (const float*)d_in[4];

    float* out   = (float*)d_out;
    float* probs = out;                                  // [B,S,E]
    float* topkf = out + (size_t)B_ * S_ * E_;           // [B,S,K] (indices as float)
    float* nbias = topkf + (size_t)B_ * S_ * K_;         // [E]

    unsigned short* wrb = (unsigned short*)d_ws;         // E*D bf16
    unsigned short* wnb = wrb + (size_t)E_ * D_;         // E*D bf16

    prep_kernel<<<(E_ * D_ + 255) / 256, 256, 0, stream>>>(Wr, Wn, bias, wrb, wnb, nbias);

    router_kernel<<<NTOK / 64, 128, 0, stream>>>(x, bias, noise_u, wrb, wnb, probs, topkf);
}
